// Model_39745627357231
// MI455X (gfx1250) — compile-verified, matrix operand order checked by reading, not run on hardware
//
#include <hip/hip_runtime.h>
#include <hip/hip_bf16.h>
#include <math.h>

// ---------------- problem constants ----------------
#define N_TOK  8192
#define DMODEL 512
#define NEXP   8
#define HDIM   1024
#define RHDIM  256
#define INDIM  4611
#define KPAD   4640     // 145 * 32, zero padded
#define KT_X   145      // KPAD/32
#define LNEPS  1e-5f

typedef unsigned short u16;
typedef unsigned int   u32;

typedef __attribute__((ext_vector_type(16))) __bf16 v16bf;
typedef __attribute__((ext_vector_type(8)))  float  v8f;

// ---------------- CDNA5 async-to-LDS path (guarded) ----------------
#if defined(__has_builtin)
#  if __has_builtin(__builtin_amdgcn_global_load_async_to_lds_b128)
#    define USE_ASYNC_LDS 1
#  endif
#endif
#ifndef USE_ASYNC_LDS
#  define USE_ASYNC_LDS 0
#endif

#if USE_ASYNC_LDS
// builtin signature (from hipcc diagnostic): arg0 = int4-vector pointer in AS(1),
// arg1 = LDS-side pointer, arg2/arg3 = imm offset / imm cpol.
typedef int v4i_b128 __attribute__((vector_size(16)));
typedef __attribute__((address_space(1))) v4i_b128* as1_v4i_ptr;
typedef __attribute__((address_space(3))) v4i_b128* as3_v4i_ptr;

__device__ inline void async_b128(const u16* g, u16* l) {
    __builtin_amdgcn_global_load_async_to_lds_b128(
        (as1_v4i_ptr)const_cast<u16*>(g),
        (as3_v4i_ptr)l,
        0, 0);
}
#if __has_builtin(__builtin_amdgcn_s_wait_asynccnt)
__device__ inline void s_wait_async0() { __builtin_amdgcn_s_wait_asynccnt(0); }
__device__ inline void s_wait_async4() { __builtin_amdgcn_s_wait_asynccnt(4); }
#else
__device__ inline void s_wait_async0() { asm volatile("s_wait_asynccnt 0x0" ::: "memory"); }
__device__ inline void s_wait_async4() { asm volatile("s_wait_asynccnt 0x4" ::: "memory"); }
#endif
#endif

// ---------------- scalar helpers ----------------
__device__ inline u16 f2bf(float f) {
    u32 u = __float_as_uint(f);
    u32 lsb = (u >> 16) & 1u;
    u += 0x7fffu + lsb;                 // round to nearest even
    return (u16)(u >> 16);
}
__device__ inline float bf2f(u16 h) {
    return __uint_as_float(((u32)h) << 16);
}
__device__ inline float gelu_exact(float x) {
    return 0.5f * x * (1.0f + erff(x * 0.70710678118654752f));
}

// block-wide sum over 256 threads
__device__ inline float blk_sum(float v, float* red) {
    int t = threadIdx.x;
    red[t] = v;
    __syncthreads();
    #pragma unroll
    for (int s = 128; s > 0; s >>= 1) {
        if (t < s) red[t] += red[t + s];
        __syncthreads();
    }
    float r = red[0];
    __syncthreads();
    return r;
}

// layernorm of a 512-vector held as 2 regs/thread (256 threads)
__device__ inline void ln_pair(float& a, float& b,
                               const float* __restrict__ g,
                               const float* __restrict__ be,
                               float* red) {
    int t = threadIdx.x;
    float m = blk_sum(a + b, red) * (1.0f / 512.0f);
    float da = a - m, db = b - m;
    float v = blk_sum(da * da + db * db, red) * (1.0f / 512.0f);
    float inv = rsqrtf(v + LNEPS);
    a = da * inv * g[t]       + be[t];
    b = db * inv * g[t + 256] + be[t + 256];
}

// ---------------- prep: build bf16 fused input X + residual ----------------
__global__ __launch_bounds__(256) void prep_kernel(
    const float* __restrict__ z1, const float* __restrict__ z2,
    const float* __restrict__ z3, const float* __restrict__ mask,
    const float* __restrict__ ln_g, const float* __restrict__ ln_b,
    const float* __restrict__ lnp_g, const float* __restrict__ lnp_b,
    u16* __restrict__ Xbf, float* __restrict__ accOut) {
    __shared__ float s1[512], s2[512], s3[512];
    __shared__ float red[256];
    const int n = blockIdx.x;
    const int t = threadIdx.x;
    const size_t off = (size_t)n * DMODEL;
    u16* xr = Xbf + (size_t)n * KPAD;

    float a1 = z1[off + t], b1 = z1[off + t + 256];
    float a2 = z2[off + t], b2 = z2[off + t + 256];
    float a3 = z3[off + t], b3 = z3[off + t + 256];
    ln_pair(a1, b1, ln_g, ln_b, red);
    ln_pair(a2, b2, ln_g, ln_b, red);
    ln_pair(a3, b3, ln_g, ln_b, red);
    s1[t] = a1; s1[t + 256] = b1;
    s2[t] = a2; s2[t + 256] = b2;
    s3[t] = a3; s3[t + 256] = b3;
    xr[t]            = f2bf(a1); xr[t + 256]        = f2bf(b1);
    xr[512 + t]      = f2bf(a2); xr[512 + t + 256]  = f2bf(b2);
    xr[1024 + t]     = f2bf(a3); xr[1024 + t + 256] = f2bf(b3);
    __syncthreads();

    const float* pa[6] = { s1, s1, s2, s1, s1, s2 };
    const float* pb[6] = { s2, s3, s3, s2, s3, s3 };
    #pragma unroll 1
    for (int c = 0; c < 6; ++c) {
        float a, b;
        if (c < 3) { a = pa[c][t] - pb[c][t]; b = pa[c][t + 256] - pb[c][t + 256]; }
        else       { a = pa[c][t] * pb[c][t]; b = pa[c][t + 256] * pb[c][t + 256]; }
        ln_pair(a, b, lnp_g, lnp_b, red);
        xr[1536 + c * 512 + t]       = f2bf(a);
        xr[1536 + c * 512 + t + 256] = f2bf(b);
    }

    // mask columns (4608..4610) + zero pad (4611..4639)
    if (t < 3)        xr[4608 + t] = f2bf(mask[(size_t)n * 3 + t]);
    else if (t < 32)  xr[4608 + t] = 0;

    // masked residual into accumulation buffer
    float m0 = mask[(size_t)n * 3], m1 = mask[(size_t)n * 3 + 1], m2 = mask[(size_t)n * 3 + 2];
    float denom = fmaxf(m0 + m1 + m2, 1.0f);
    accOut[off + t]       = (m0 * s1[t]       + m1 * s2[t]       + m2 * s3[t])       / denom;
    accOut[off + t + 256] = (m0 * s1[t + 256] + m1 * s2[t + 256] + m2 * s3[t + 256]) / denom;
}

// ---------------- weight pack: fp32 [E,K,Ncols] -> bf16 transposed [E,Ncols,Kpad] ----------------
__global__ __launch_bounds__(256) void pack_bt_kernel(
    const float* __restrict__ w, u16* __restrict__ wt,
    int E_, int K, int Ncols, int Kpad) {
    size_t idx = (size_t)blockIdx.x * blockDim.x + threadIdx.x;
    size_t total = (size_t)E_ * Ncols * Kpad;
    if (idx >= total) return;
    int k = (int)(idx % Kpad);
    int n = (int)((idx / Kpad) % Ncols);
    int e = (int)(idx / ((size_t)Kpad * Ncols));
    u16 v = 0;
    if (k < K) v = f2bf(w[((size_t)e * K + k) * Ncols + n]);
    wt[idx] = v;
}

// ---------------- WMMA GEMM ----------------
// C[M,Ncols] = A[M,Kpad](bf16) * Bt[Ncols,Kpad](bf16)^T, f32 accumulate.
// Block tile 128x128, K-tile 32, 8 waves: wave = 64x32 region = 4x2 fragments.
#define BM 128
#define BN 128
#define BK 32

__device__ inline v16bf load_frag16(const u16* base, int half) {
    union { uint4 q[2]; v16bf v; } u;
    u.q[0] = *(const uint4*)(base + half * 8);        // K group {h*8 .. h*8+7}
    u.q[1] = *(const uint4*)(base + 16 + half * 8);   // K group {16+h*8 .. 16+h*8+7}
    return u.v;
}

__device__ inline void do_tile(const u16* Asb, const u16* Bsb,
                               int wm, int wn, int l16, int half,
                               v8f (&acc)[4][2]) {
    v16bf bfr[2];
    #pragma unroll
    for (int j = 0; j < 2; ++j) {
        int c = wn * 32 + j * 16 + l16;
        bfr[j] = load_frag16(Bsb + c * BK, half);
    }
    #pragma unroll
    for (int i = 0; i < 4; ++i) {
        int r = wm * 64 + i * 16 + l16;
        v16bf afr = load_frag16(Asb + r * BK, half);
        #pragma unroll
        for (int j = 0; j < 2; ++j) {
            acc[i][j] = __builtin_amdgcn_wmma_f32_16x16x32_bf16(
                false, afr, false, bfr[j], (short)0, acc[i][j], false, false);
        }
    }
}

// EPI 0: out_bf = bf16(gelu(acc + bias[col]))
// EPI 1: out_f32[row*ldOut+col] += gate[row*NEXP+expert] * (acc + bias[col])
template <int EPI>
__global__ __launch_bounds__(256) void wmma_gemm_kernel(
    const u16* __restrict__ A, const u16* __restrict__ Bt,
    int Kpad, int Ktiles,
    const float* __restrict__ bias,
    const float* __restrict__ gate, int expertIdx,
    u16* __restrict__ outBf, float* __restrict__ outF, int ldOut) {
#if USE_ASYNC_LDS
    __shared__ alignas(16) u16 As[2][BM * BK];
    __shared__ alignas(16) u16 Bs[2][BN * BK];
#else
    __shared__ alignas(16) u16 As[1][BM * BK];
    __shared__ alignas(16) u16 Bs[1][BN * BK];
#endif

    const int tid  = threadIdx.x;
    const int lane = tid & 31;
    const int wid  = tid >> 5;
    const int wm   = wid >> 2;     // 0..1
    const int wn   = wid & 3;      // 0..3
    const int half = lane >> 4;    // K-group select
    const int l16  = lane & 15;    // row (A) / col (B,C)
    const int rowBase = blockIdx.y * BM;
    const int colBase = blockIdx.x * BN;

    v8f acc[4][2];
    const v8f vzero = {0.f, 0.f, 0.f, 0.f, 0.f, 0.f, 0.f, 0.f};
    #pragma unroll
    for (int i = 0; i < 4; ++i)
        #pragma unroll
        for (int j = 0; j < 2; ++j) acc[i][j] = vzero;

#if USE_ASYNC_LDS
    // 4 async b128 per thread per tile (2 for A, 2 for B) -> tracked by ASYNCcnt
    auto issue = [&](int kt, u16* Ad, u16* Bd) {
        const int k0 = kt * BK;
        #pragma unroll
        for (int it = 0; it < 2; ++it) {
            int i   = tid + it * 256;
            int r   = i >> 2;
            int seg = i & 3;
            async_b128(A  + (size_t)(rowBase + r) * Kpad + k0 + seg * 8,
                       Ad + r * BK + seg * 8);
            async_b128(Bt + (size_t)(colBase + r) * Kpad + k0 + seg * 8,
                       Bd + r * BK + seg * 8);
        }
    };
    issue(0, As[0], Bs[0]);
    for (int kt = 0; kt < Ktiles; ++kt) {
        const int cur = kt & 1;
        if (kt + 1 < Ktiles) {
            issue(kt + 1, As[cur ^ 1], Bs[cur ^ 1]);  // overlap DMA with compute
            s_wait_async4();                          // tile kt has landed
        } else {
            s_wait_async0();
        }
        __syncthreads();
        do_tile(As[cur], Bs[cur], wm, wn, l16, half, acc);
        __syncthreads();
    }
#else
    for (int kt = 0; kt < Ktiles; ++kt) {
        const int k0 = kt * BK;
        #pragma unroll
        for (int it = 0; it < 2; ++it) {
            int i   = tid + it * 256;
            int r   = i >> 2;
            int seg = i & 3;
            *(uint4*)(As[0] + r * BK + seg * 8) =
                *(const uint4*)(A  + (size_t)(rowBase + r) * Kpad + k0 + seg * 8);
            *(uint4*)(Bs[0] + r * BK + seg * 8) =
                *(const uint4*)(Bt + (size_t)(colBase + r) * Kpad + k0 + seg * 8);
        }
        __syncthreads();
        do_tile(As[0], Bs[0], wm, wn, l16, half, acc);
        __syncthreads();
    }
#endif

    // epilogue; C/D layout: VGPR r -> row r + 8*(lane/16), col = lane%16
    #pragma unroll
    for (int i = 0; i < 4; ++i) {
        #pragma unroll
        for (int j = 0; j < 2; ++j) {
            const int colG = colBase + wn * 32 + j * 16 + l16;
            const float bcol = bias[colG];
            #pragma unroll
            for (int r = 0; r < 8; ++r) {
                const int rowG = rowBase + wm * 64 + i * 16 + half * 8 + r;
                float v = acc[i][j][r] + bcol;
                if (EPI == 0) {
                    outBf[(size_t)rowG * ldOut + colG] = f2bf(gelu_exact(v));
                } else {
                    float gt = gate[(size_t)rowG * NEXP + expertIdx];
                    outF[(size_t)rowG * ldOut + colG] += gt * v;
                }
            }
        }
    }
}

// ---------------- router tail: 256->8, temperature, top-2 softmax ----------------
__global__ __launch_bounds__(256) void router_gate_kernel(
    const u16* __restrict__ RH_, const float* __restrict__ w2,
    const float* __restrict__ b2, const float* __restrict__ logTemp,
    float* __restrict__ gate) {
    __shared__ float lg[32][NEXP];
    const int tid = threadIdx.x;
    const int e  = tid & 7;
    const int lt = tid >> 3;                 // 0..31 tokens per block
    const int tok = blockIdx.x * 32 + lt;

    float sum = b2[e];
    const u16* rh = RH_ + (size_t)tok * RHDIM;
    for (int i = 0; i < RHDIM; ++i) sum += bf2f(rh[i]) * w2[i * NEXP + e];
    float temp = expf(logTemp[0]);
    temp = fminf(fmaxf(temp, 1e-3f), 100.0f);
    lg[lt][e] = sum / temp;
    __syncthreads();

    if (e == 0) {
        float v[NEXP];
        #pragma unroll
        for (int i = 0; i < NEXP; ++i) v[i] = lg[lt][i];
        int i1 = 0; float m1 = v[0];
        #pragma unroll
        for (int i = 1; i < NEXP; ++i) if (v[i] > m1) { m1 = v[i]; i1 = i; }
        int i2 = -1; float m2 = -INFINITY;
        #pragma unroll
        for (int i = 0; i < NEXP; ++i) if (i != i1 && v[i] > m2) { m2 = v[i]; i2 = i; }
        float e2 = expf(m2 - m1);
        float s = 1.0f + e2;
        float* g = gate + (size_t)tok * NEXP;
        #pragma unroll
        for (int i = 0; i < NEXP; ++i) g[i] = 0.f;
        g[i1] = 1.0f / s;
        g[i2] = e2 / s;
    }
}

// ---------------- output layernorm ----------------
__global__ __launch_bounds__(256) void final_ln_kernel(
    const float* __restrict__ accIn, const float* __restrict__ g,
    const float* __restrict__ be, float* __restrict__ out) {
    __shared__ float red[256];
    const int n = blockIdx.x;
    const int t = threadIdx.x;
    const size_t off = (size_t)n * DMODEL;
    float a = accIn[off + t], b = accIn[off + t + 256];
    ln_pair(a, b, g, be, red);
    out[off + t] = a;
    out[off + t + 256] = b;
}

// ---------------- orchestration ----------------
extern "C" void kernel_launch(void* const* d_in, const int* in_sizes, int n_in,
                              void* d_out, int out_size, void* d_ws, size_t ws_size,
                              hipStream_t stream) {
    (void)in_sizes; (void)n_in; (void)out_size; (void)ws_size;
    const float* z1    = (const float*)d_in[0];
    const float* z2    = (const float*)d_in[1];
    const float* z3    = (const float*)d_in[2];
    const float* mask  = (const float*)d_in[3];
    const float* ln_g  = (const float*)d_in[4];
    const float* ln_b  = (const float*)d_in[5];
    const float* lnp_g = (const float*)d_in[6];
    const float* lnp_b = (const float*)d_in[7];
    const float* oln_g = (const float*)d_in[8];
    const float* oln_b = (const float*)d_in[9];
    const float* r_w1  = (const float*)d_in[10];
    const float* r_b1  = (const float*)d_in[11];
    const float* r_w2  = (const float*)d_in[12];
    const float* r_b2  = (const float*)d_in[13];
    const float* logT  = (const float*)d_in[14];
    const float* e_w1  = (const float*)d_in[15];
    const float* e_b1  = (const float*)d_in[16];
    const float* e_w2  = (const float*)d_in[17];
    const float* e_b2  = (const float*)d_in[18];
    float* out = (float*)d_out;

    // workspace carve (~201 MB total)
    char* w = (char*)d_ws;
    auto carve = [&](size_t bytes) {
        void* p = (void*)w;
        w += (bytes + 255) & ~(size_t)255;
        return p;
    };
    u16*   Xbf   = (u16*)  carve((size_t)N_TOK * KPAD * 2);           // 76 MB
    u16*   W1T   = (u16*)  carve((size_t)NEXP * HDIM * KPAD * 2);     // 76 MB
    u16*   RW1T  = (u16*)  carve((size_t)RHDIM * KPAD * 2);           // 2.4 MB
    u16*   W2T   = (u16*)  carve((size_t)NEXP * DMODEL * HDIM * 2);   // 8.4 MB
    u16*   Hbuf  = (u16*)  carve((size_t)N_TOK * HDIM * 2);           // 16.8 MB
    u16*   RHbuf = (u16*)  carve((size_t)N_TOK * RHDIM * 2);          // 4.2 MB
    float* gate  = (float*)carve((size_t)N_TOK * NEXP * 4);           // 0.26 MB
    float* accB  = (float*)carve((size_t)N_TOK * DMODEL * 4);         // 16.8 MB

    // 1) fused-input build + residual seed
    prep_kernel<<<N_TOK, 256, 0, stream>>>(z1, z2, z3, mask, ln_g, ln_b,
                                           lnp_g, lnp_b, Xbf, accB);

    // 2) weight conversions (fp32 -> transposed bf16, K zero-padded)
    {
        size_t tot = (size_t)NEXP * HDIM * KPAD;
        pack_bt_kernel<<<(unsigned)((tot + 255) / 256), 256, 0, stream>>>(
            e_w1, W1T, NEXP, INDIM, HDIM, KPAD);
        tot = (size_t)RHDIM * KPAD;
        pack_bt_kernel<<<(unsigned)((tot + 255) / 256), 256, 0, stream>>>(
            r_w1, RW1T, 1, INDIM, RHDIM, KPAD);
        tot = (size_t)NEXP * DMODEL * HDIM;
        pack_bt_kernel<<<(unsigned)((tot + 255) / 256), 256, 0, stream>>>(
            e_w2, W2T, NEXP, HDIM, DMODEL, HDIM);
    }

    // 3) router fc1 (8192 x 4640 x 256) + gating
    wmma_gemm_kernel<0><<<dim3(RHDIM / BN, N_TOK / BM), 256, 0, stream>>>(
        Xbf, RW1T, KPAD, KT_X, r_b1, nullptr, 0, RHbuf, nullptr, RHDIM);
    router_gate_kernel<<<N_TOK / 32, 256, 0, stream>>>(RHbuf, r_w2, r_b2, logT, gate);

    // 4) experts, serialized so Hbuf is reused
    for (int e = 0; e < NEXP; ++e) {
        wmma_gemm_kernel<0><<<dim3(HDIM / BN, N_TOK / BM), 256, 0, stream>>>(
            Xbf, W1T + (size_t)e * HDIM * KPAD, KPAD, KT_X,
            e_b1 + (size_t)e * HDIM, nullptr, 0, Hbuf, nullptr, HDIM);
        wmma_gemm_kernel<1><<<dim3(DMODEL / BN, N_TOK / BM), 256, 0, stream>>>(
            Hbuf, W2T + (size_t)e * DMODEL * HDIM, HDIM, HDIM / BK,
            e_b2 + (size_t)e * DMODEL, gate, e, nullptr, accB, DMODEL);
    }

    // 5) output layernorm
    final_ln_kernel<<<N_TOK, 256, 0, stream>>>(accB, oln_g, oln_b, out);
}
